// Encoder_38852274159790
// MI455X (gfx1250) — matrix operand order repaired and checked by reading, not compiled, not run on hardware
//
#include <hip/hip_runtime.h>
#include <hip/hip_bf16.h>

typedef __attribute__((ext_vector_type(16))) _Float16 v16h;
typedef __attribute__((ext_vector_type(8)))  _Float16 v8h;
typedef __attribute__((ext_vector_type(8)))  float    v8f;

#define BATCH 32
#define SEQ   128
#define UNITS 512
#define N3    1536   // 3*UNITS
#define EMB   512

// ---------------------------------------------------------------------------
// WMMA fragment loaders for v_wmma_f32_16x16x32_f16 (A and B both use contiguous k = 16*(L/16) + e)
// ---------------------------------------------------------------------------

// A fragment: 16x32 f16 tile from row-major matrix, row stride ld (halves).
// lane L: row = L%16; K-halves: base=(L/16)*16 -> elems 0..7 = K base..base+7,
// elems 8..15 = K base+8..base+15.
__device__ inline v16h load_frag_a(const _Float16* base, int ld) {
    int lane = threadIdx.x & 31;
    int m    = lane & 15;
    int kg   = (lane >> 4) * 16;
    const _Float16* p = base + (size_t)m * ld + kg;
    v8h lo = *(const v8h*)(p);
    v8h hi = *(const v8h*)(p + 8);
    v16h r;
#pragma unroll
    for (int i = 0; i < 8; ++i) { r[i] = lo[i]; r[8 + i] = hi[i]; }
    return r;
}

// B fragment: 32x16 f16 tile; weights pre-transposed [N][K] so a B column is a
// contiguous row. lane L: n = L%16, K = (L/16)*16 + e (e=0..15).
__device__ inline v16h load_frag_b(const _Float16* baseT, int ld) {
    int lane = threadIdx.x & 31;
    int n    = lane & 15;
    int kg   = (lane >> 4) * 16;
    return *(const v16h*)(baseT + (size_t)n * ld + kg);
}

// ---------------------------------------------------------------------------
// Weight prep: W [K][N] f32 -> Wt [N][K] f16
// ---------------------------------------------------------------------------
__global__ void wprep_kernel(const float* __restrict__ W, _Float16* __restrict__ Wt,
                             int K, int N) {
    int i = blockIdx.x * blockDim.x + threadIdx.x;
    if (i >= K * N) return;
    int n = i % N;
    int k = i / N;
    Wt[(size_t)n * K + k] = (_Float16)W[i];
}

// ---------------------------------------------------------------------------
// Embedding gather: e_h[b,t,:] = f16(E[x[b,t],:])
// ---------------------------------------------------------------------------
__global__ void embed_kernel(const int* __restrict__ x, const float* __restrict__ E,
                             _Float16* __restrict__ eh) {
    int i = blockIdx.x * blockDim.x + threadIdx.x;     // over BATCH*SEQ*EMB
    if (i >= BATCH * SEQ * EMB) return;
    int bt = i >> 9;          // /EMB
    int c  = i & (EMB - 1);
    int tok = x[bt];
    eh[i] = (_Float16)E[(size_t)tok * EMB + c];
}

// ---------------------------------------------------------------------------
// Input projection GEMM: C[M,N] f32 = A[M,K] f16 * Bt[N,K]^T f16 + bias[N]
// One wave per 32x64 output block: 2 M-tiles x 4 N-tiles, so every B fragment
// feeds 2 WMMAs and every A fragment feeds 4 (8 WMMA / K-chunk / wave).
// ---------------------------------------------------------------------------
__global__ __launch_bounds__(256)
void gemm_xproj_kernel(const _Float16* __restrict__ A, const _Float16* __restrict__ Bt,
                       const float* __restrict__ bias, float* __restrict__ C,
                       int M, int N, int K) {
    int wave = (blockIdx.x * blockDim.x + threadIdx.x) >> 5;
    int ntg  = N >> 6;                   // strips of 64 columns
    int mt   = wave / ntg;               // block of 32 rows
    int ng   = wave % ntg;
    if (mt >= (M >> 5)) return;

    v8f acc0[4], acc1[4];
#pragma unroll
    for (int j = 0; j < 4; ++j) {
        acc0[j] = (v8f){0.f,0.f,0.f,0.f,0.f,0.f,0.f,0.f};
        acc1[j] = (v8f){0.f,0.f,0.f,0.f,0.f,0.f,0.f,0.f};
    }

    const _Float16* A0 = A + (size_t)(mt * 32) * K;
    const _Float16* A1 = A0 + (size_t)16 * K;
    for (int kt = 0; kt < K; kt += 32) {
        v16h a0 = load_frag_a(A0 + kt, K);
        v16h a1 = load_frag_a(A1 + kt, K);
#pragma unroll
        for (int j = 0; j < 4; ++j) {
            const _Float16* bb = Bt + (size_t)((ng * 4 + j) * 16) * K + kt;
            v16h b = load_frag_b(bb, K);
            acc0[j] = __builtin_amdgcn_wmma_f32_16x16x32_f16(
                false, a0, false, b, (short)0, acc0[j], false, false);
            acc1[j] = __builtin_amdgcn_wmma_f32_16x16x32_f16(
                false, a1, false, b, (short)0, acc1[j], false, false);
        }
    }

    int lane = threadIdx.x & 31;
    int n    = lane & 15;
    int mg   = (lane >> 4) * 8;
#pragma unroll
    for (int j = 0; j < 4; ++j) {
        int ncol = (ng * 4 + j) * 16 + n;
        float bv = bias[ncol];
        float* cb0 = C + (size_t)(mt * 32) * N + ncol;
        float* cb1 = cb0 + (size_t)16 * N;
#pragma unroll
        for (int r = 0; r < 8; ++r) {
            cb0[(size_t)(mg + r) * N] = acc0[j][r] + bv;
            cb1[(size_t)(mg + r) * N] = acc1[j][r] + bv;
        }
    }
}

// ---------------------------------------------------------------------------
// Persistent GRU recurrence over T steps; 16 batch rows per workgroup
// (blockIdx.x), optional second direction on blockIdx.y so fw/bw of the
// bidirectional layer run concurrently on separate WGPs.
// 16 waves compute rec = h @ U^T via WMMA (h f16 in LDS), then all 512 threads
// do the gate math in f32 with the running state carried in registers.
// Keras GRU reset_after=True, both activations sigmoid, masked steps carry h.
// ---------------------------------------------------------------------------
__device__ inline float sigmoidf(float v) { return 1.0f / (1.0f + __expf(-v)); }

__global__ __launch_bounds__(512)
void gru_recur_kernel(const float* __restrict__ xp0, const float* __restrict__ xp1,
                      const _Float16* __restrict__ Ut0, const _Float16* __restrict__ Ut1,
                      const float* __restrict__ br0, const float* __restrict__ br1,
                      const int* __restrict__ x,        // [BATCH,T] tokens (mask)
                      _Float16* __restrict__ yh,        // f16 output base
                      int yld,                          // row stride (halves)
                      float* __restrict__ yf32,         // optional f32 out [BATCH,T,UNITS]
                      int T) {
    __shared__ _Float16 hbuf[16 * UNITS];   // 16 KB
    __shared__ _Float16 rec[16 * N3];       // 48 KB

    const int dir = blockIdx.y;             // 0 = fw, 1 = bw (when gridDim.y==2)
    const float*    xp   = dir ? xp1 : xp0;
    const _Float16* Ut   = dir ? Ut1 : Ut0;
    const float*    brec = dir ? br1 : br0;
    const int yoff = dir * UNITS;           // bw writes concat columns 512..1023
    const int reverse = dir;

    const int b0   = blockIdx.x * 16;
    const int tid  = threadIdx.x;
    const int wave = tid >> 5;
    const int lane = tid & 31;

    for (int i = tid; i < 16 * UNITS; i += 512) hbuf[i] = (_Float16)0.f;
    float hreg[16];
#pragma unroll
    for (int j = 0; j < 16; ++j) hreg[j] = 0.f;
    __syncthreads();

    for (int s = 0; s < T; ++s) {
        const int t = reverse ? (T - 1 - s) : s;

        // ---- rec = h @ U^T + b_rec  (M=16, N=1536, K=512) ----
        v8f acc[6];
#pragma unroll
        for (int j = 0; j < 6; ++j) acc[j] = (v8f){0.f,0.f,0.f,0.f,0.f,0.f,0.f,0.f};

        for (int kt = 0; kt < UNITS; kt += 32) {
            v16h a = load_frag_a(hbuf + kt, UNITS);
#pragma unroll
            for (int j = 0; j < 6; ++j) {
                int nt = wave * 6 + j;
                v16h b = load_frag_b(Ut + (size_t)(nt * 16) * UNITS + kt, UNITS);
                acc[j] = __builtin_amdgcn_wmma_f32_16x16x32_f16(
                    false, a, false, b, (short)0, acc[j], false, false);
            }
        }
        {
            int n  = lane & 15;
            int mg = (lane >> 4) * 8;
#pragma unroll
            for (int j = 0; j < 6; ++j) {
                int ncol = (wave * 6 + j) * 16 + n;
                float bv = brec[ncol];
#pragma unroll
                for (int r = 0; r < 8; ++r)
                    rec[(mg + r) * N3 + ncol] = (_Float16)(acc[j][r] + bv);
            }
        }
        __syncthreads();

        // ---- gating: thread tid owns unit u=tid across all 16 batch rows ----
        const int u = tid;
#pragma unroll
        for (int j = 0; j < 16; ++j) {
            int gb = b0 + j;
            const float* xpr = xp + ((size_t)gb * T + t) * N3;
            float xz = xpr[u];
            float xr = xpr[UNITS + u];
            float xh = xpr[2 * UNITS + u];
            float rz = (float)rec[j * N3 + u];
            float rr = (float)rec[j * N3 + UNITS + u];
            float rh = (float)rec[j * N3 + 2 * UNITS + u];
            float h  = hreg[j];
            float z  = sigmoidf(xz + rz);
            float r  = sigmoidf(xr + rr);
            float hh = sigmoidf(xh + r * rh);
            float hn = z * h + (1.f - z) * hh;
            bool mk  = (x[(size_t)gb * T + t] != 0);
            hn = mk ? hn : h;
            hreg[j] = hn;
            _Float16 hf = (_Float16)hn;
            hbuf[j * UNITS + u] = hf;
            yh[((size_t)gb * T + t) * yld + yoff + u] = hf;
            if (yf32) yf32[((size_t)gb * T + t) * UNITS + u] = hn;
        }
        __syncthreads();
    }
}

// ---------------------------------------------------------------------------
// hT = out[:, T-1, :]   (masked steps carry state, so last y == last state)
// ---------------------------------------------------------------------------
__global__ void copy_hT_kernel(const float* __restrict__ y, float* __restrict__ o, int T) {
    int i = blockIdx.x * blockDim.x + threadIdx.x;      // BATCH*UNITS
    if (i >= BATCH * UNITS) return;
    int b = i >> 9;
    int u = i & (UNITS - 1);
    o[i] = y[((size_t)b * T + (T - 1)) * UNITS + u];
}

// ---------------------------------------------------------------------------
extern "C" void kernel_launch(void* const* d_in, const int* in_sizes, int n_in,
                              void* d_out, int out_size, void* d_ws, size_t ws_size,
                              hipStream_t stream) {
    (void)in_sizes; (void)n_in; (void)out_size; (void)ws_size;

    const int*   x    = (const int*)d_in[0];
    // d_in[1] = hidden (unused by the reference body)
    const float* E    = (const float*)d_in[2];
    const float* W[5] = { (const float*)d_in[3],  (const float*)d_in[6],
                          (const float*)d_in[9],  (const float*)d_in[12],
                          (const float*)d_in[15] };
    const float* U[5] = { (const float*)d_in[4],  (const float*)d_in[7],
                          (const float*)d_in[10], (const float*)d_in[13],
                          (const float*)d_in[16] };
    const float* Bb[5]= { (const float*)d_in[5],  (const float*)d_in[8],
                          (const float*)d_in[11], (const float*)d_in[14],
                          (const float*)d_in[17] };
    const int Kin[5]  = { EMB, EMB, 2 * UNITS, UNITS, UNITS };

    // ---- workspace bump allocator ----
    char* ws = (char*)d_ws;
    size_t off = 0;
    auto alloc = [&](size_t bytes) -> char* {
        char* p = ws + off;
        off = (off + bytes + 255) & ~(size_t)255;
        return p;
    };
    _Float16* e_h   = (_Float16*)alloc((size_t)BATCH * SEQ * EMB * 2);
    _Float16* Wt[5];
    for (int l = 0; l < 5; ++l) Wt[l] = (_Float16*)alloc((size_t)Kin[l] * N3 * 2);
    _Float16* Ut[5];
    for (int l = 0; l < 5; ++l) Ut[l] = (_Float16*)alloc((size_t)UNITS * N3 * 2);
    float*    xpA   = (float*)alloc((size_t)BATCH * SEQ * N3 * 4);
    float*    xpB   = (float*)alloc((size_t)BATCH * SEQ * N3 * 4);
    _Float16* ycat  = (_Float16*)alloc((size_t)BATCH * SEQ * (2 * UNITS) * 2);
    _Float16* y1    = (_Float16*)alloc((size_t)BATCH * SEQ * UNITS * 2);
    _Float16* y2    = (_Float16*)alloc((size_t)BATCH * SEQ * UNITS * 2);

    const int BT = BATCH * SEQ;   // 4096 rows in the projection GEMMs

    // ---- 1) weight prep: transpose + f32->f16 ----
    for (int l = 0; l < 5; ++l) {
        int nW = Kin[l] * N3;
        wprep_kernel<<<(nW + 255) / 256, 256, 0, stream>>>(W[l], Wt[l], Kin[l], N3);
        int nU = UNITS * N3;
        wprep_kernel<<<(nU + 255) / 256, 256, 0, stream>>>(U[l], Ut[l], UNITS, N3);
    }

    // ---- 2) embedding gather ----
    {
        int n = BT * EMB;
        embed_kernel<<<(n + 255) / 256, 256, 0, stream>>>(x, E, e_h);
    }

    auto launch_gemm = [&](const _Float16* A, int l, float* xpdst) {
        int waves  = (BT / 32) * (N3 / 64);
        int blocks = (waves * 32) / 256;
        gemm_xproj_kernel<<<blocks, 256, 0, stream>>>(A, Wt[l], Bb[l], xpdst,
                                                      BT, N3, Kin[l]);
    };
    auto launch_recur_uni = [&](int l, const float* xpsrc, _Float16* yout, int yld,
                                float* yf32) {
        gru_recur_kernel<<<dim3(BATCH / 16, 1), 512, 0, stream>>>(
            xpsrc, xpsrc, Ut[l], Ut[l], Bb[l] + N3, Bb[l] + N3,
            x, yout, yld, yf32, SEQ);
    };

    // ---- 3) bidirectional layer: both projections, then fw+bw concurrently ----
    launch_gemm(e_h, 0, xpA);
    launch_gemm(e_h, 1, xpB);
    gru_recur_kernel<<<dim3(BATCH / 16, 2), 512, 0, stream>>>(
        xpA, xpB, Ut[0], Ut[1], Bb[0] + N3, Bb[1] + N3,
        x, ycat, 2 * UNITS, nullptr, SEQ);

    // ---- 4) m0 (input = concat, K=1024) ----
    launch_gemm(ycat, 2, xpA);
    launch_recur_uni(2, xpA, y1, UNITS, nullptr);
    // ---- 5) m1 ----
    launch_gemm(y1, 3, xpA);
    launch_recur_uni(3, xpA, y2, UNITS, nullptr);
    // ---- 6) top (f32 output straight to d_out) ----
    launch_gemm(y2, 4, xpA);
    float* out_y = (float*)d_out;
    launch_recur_uni(4, xpA, y1 /*scratch*/, UNITS, out_y);

    // ---- 7) hT = out[:, T-1, :] ----
    float* out_hT = out_y + (size_t)BATCH * SEQ * UNITS;
    copy_hT_kernel<<<(BATCH * UNITS + 255) / 256, 256, 0, stream>>>(out_y, out_hT, SEQ);
}
